// ScaledDotProductAttention_32298154066178
// MI455X (gfx1250) — compile-verified, matrix operand order checked by reading, not run on hardware
//
#include <hip/hip_runtime.h>

typedef __attribute__((ext_vector_type(16))) __bf16 v16bf;
typedef __attribute__((ext_vector_type(8)))  __bf16 v8bf;
typedef __attribute__((ext_vector_type(8)))  float  v8f;

#define S_LEN 2048
#define D_DIM 64
#define QTILE 16
#define BH_N  16                      // B*H
#define SCALE 0.125f                  // 1/sqrt(64)
#define SMEM_BYTES (QTILE * S_LEN * 4)
#define N_ELEM ((long)BH_N * S_LEN * D_DIM)         // 2,097,152
#define WS_NEEDED ((size_t)N_ELEM * 2 * 3)          // Qb + Kb + Vt (bf16)

// K-index mapping for 16-bit A/B fragments (ISA 7.12.2):
// lanes 0-15 hold K = {0..7, 16..23}, lanes 16-31 hold K = {8..15, 24..31}
__device__ __forceinline__ int kmap(int e, int h) {
    return (e < 8) ? (8 * h + e) : (16 + 8 * h + (e - 8));
}

// round-to-nearest-even f32 -> bf16 (bit-exact, compile-safe)
__device__ __forceinline__ __bf16 f2bf(float f) {
    union { float f; unsigned u; } x; x.f = f;
    unsigned short h = (unsigned short)((x.u + 0x7FFFu + ((x.u >> 16) & 1u)) >> 16);
    __bf16 b;
    __builtin_memcpy(&b, &h, 2);
    return b;
}
__device__ __forceinline__ unsigned bf16bits(float f) {
    union { float f; unsigned u; } x; x.f = f;
    return (x.u + 0x7FFFu + ((x.u >> 16) & 1u)) >> 16;
}

__device__ __forceinline__ float wave_sum(float v) {
#pragma unroll
    for (int off = 16; off > 0; off >>= 1) v += __shfl_xor(v, off, 32);
    return v;
}

// assemble a 16-element fragment from two contiguous 8-element groups
__device__ __forceinline__ v16bf frag_g(const __bf16* base, int d0) {
    v8bf g0 = *(const v8bf*)(base + d0);
    v8bf g1 = *(const v8bf*)(base + d0 + 16);
    return __builtin_shufflevector(g0, g1, 0,1,2,3,4,5,6,7,8,9,10,11,12,13,14,15);
}

// ---------------- prep: f32 -> bf16 copies in workspace ----------------
__global__ __launch_bounds__(256)
void prep_kernel(const float* __restrict__ Q, const float* __restrict__ K,
                 const float* __restrict__ V,
                 __bf16* __restrict__ Qb, __bf16* __restrict__ Kb,
                 __bf16* __restrict__ Vt) {
    long i = (long)blockIdx.x * 256 + threadIdx.x;   // 0 .. N_ELEM-1
    Qb[i] = f2bf(Q[i] * SCALE);                      // fold the 1/sqrt(dk) scale
    Kb[i] = f2bf(K[i]);
    // Vt[bh][d][s] = V[bh][s][d]
    long bh  = i >> 17;                              // / (64*2048)
    long rem = i & 131071;
    long d   = rem >> 11;                            // / 2048
    long s   = rem & 2047;
    Vt[i] = f2bf(V[(bh * S_LEN + s) * D_DIM + d]);
}

// ---------------- main: WMMA attention with sparsemax ----------------
__global__ __launch_bounds__(256)
void sparsemax_attn_kernel(const __bf16* __restrict__ Qb,
                           const __bf16* __restrict__ Kb,
                           const __bf16* __restrict__ Vt,
                           float* __restrict__ O) {
    extern __shared__ float smem[];          // 16 x 2048 f32 score stripe
    const int qtile = blockIdx.x;            // 0..127
    const int bh    = blockIdx.y;            // 0..15
    const int tid   = threadIdx.x;
    const int wave  = tid >> 5;
    const int lane  = tid & 31;
    const int half  = lane >> 4;
    const int l16   = lane & 15;

    const long baseQ = ((long)bh * S_LEN + (long)qtile * QTILE) * D_DIM;
    const long baseK = (long)bh * S_LEN * D_DIM;

    // ---------- Phase 1: scores = (Q*scale) K^T via bf16 WMMA ----------
    v16bf aq[2];
#pragma unroll
    for (int c = 0; c < 2; ++c)
        aq[c] = frag_g(Qb + baseQ + l16 * D_DIM, 32 * c + 8 * half);

    for (int i = 0; i < 16; ++i) {
        int kt  = wave * 16 + i;             // key tile 0..127
        int key = kt * 16 + l16;             // B column n = l16 -> key row
        if (i < 15)                          // gfx1250 global_prefetch_b8
            __builtin_prefetch(Kb + baseK + (long)(key + 16) * D_DIM, 0, 3);
        v8f acc = {};
#pragma unroll
        for (int c = 0; c < 2; ++c) {
            v16bf bk = frag_g(Kb + baseK + (long)key * D_DIM, 32 * c + 8 * half);
            acc = __builtin_amdgcn_wmma_f32_16x16x32_bf16(
                      false, aq[c], false, bk, (short)0, acc, false, false);
        }
#pragma unroll
        for (int r = 0; r < 8; ++r) {
            int m = r + 8 * half;
            smem[m * S_LEN + kt * 16 + l16] = acc[r];
        }
    }
    __syncthreads();

    // ---------- Phase 2: sparsemax threshold (Michelot), pack P->bf16 ----------
#pragma unroll
    for (int rr = 0; rr < 2; ++rr) {
        int m = wave * 2 + rr;               // 8 waves x 2 rows
        float* row = smem + m * S_LEN;
        float s = 0.f;
        for (int i = lane; i < S_LEN; i += 32) s += row[i];
        s = wave_sum(s);
        float cnt = (float)S_LEN;
        float tau = (s - 1.f) / cnt;
        for (int it = 0; it < 24; ++it) {
            float s2 = 0.f, c2 = 0.f;
            for (int i = lane; i < S_LEN; i += 32) {
                float z = row[i];
                if (z > tau) { s2 += z; c2 += 1.f; }
            }
            s2 = wave_sum(s2);
            c2 = wave_sum(c2);
            if (c2 == cnt) break;            // support fixed -> tau exact
            cnt = c2;
            tau = (s2 - 1.f) / c2;
        }
        // pack p = max(z - tau, 0) as bf16 into the first half of the row.
        // read word 2i (>= write word i): same-wave LDS ops stay in order -> safe.
        unsigned* rowp = (unsigned*)row;
        for (int i = lane; i < S_LEN / 2; i += 32) {
            float p0 = fmaxf(row[2 * i]     - tau, 0.f);
            float p1 = fmaxf(row[2 * i + 1] - tau, 0.f);
            rowp[i] = bf16bits(p0) | (bf16bits(p1) << 16);
        }
    }
    __syncthreads();

    // ---------- Phase 3: O = P V via bf16 WMMA ----------
    v8f acc[4] = { {}, {}, {}, {} };
    const long baseV = (long)bh * D_DIM * S_LEN;     // Vt is [bh][64][2048]
    for (int j = 0; j < 8; ++j) {
        int keybase = (wave + 8 * j) * 32;           // contraction chunk
        v16bf ap = frag_g((const __bf16*)(smem + l16 * S_LEN), keybase + 8 * half);
#pragma unroll
        for (int t = 0; t < 4; ++t) {                // 4 N-tiles cover D=64
            int d = t * 16 + l16;
            v16bf bv = frag_g(Vt + baseV + (long)d * S_LEN, keybase + 8 * half);
            acc[t] = __builtin_amdgcn_wmma_f32_16x16x32_bf16(
                         false, ap, false, bv, (short)0, acc[t], false, false);
        }
    }
    __syncthreads();                                  // all waves done reading P
#pragma unroll
    for (int t = 0; t < 4; ++t)
#pragma unroll
        for (int r = 0; r < 8; ++r)
            smem[((wave * 4 + t) * 8 + r) * 32 + lane] = acc[t][r];
    __syncthreads();

    if (wave < 4) {                                   // wave t reduces N-tile t
        int t = wave;
#pragma unroll
        for (int r = 0; r < 8; ++r) {
            float s = 0.f;
#pragma unroll
            for (int w = 0; w < 8; ++w)
                s += smem[((w * 4 + t) * 8 + r) * 32 + lane];
            int mq = r + 8 * half;
            int d  = t * 16 + l16;
            long q = (long)bh * S_LEN + (long)qtile * QTILE + mq;
            O[q * D_DIM + d] = s;
        }
    }
}

// ---------------- fallback (no workspace): in-kernel conversion ----------------
__global__ __launch_bounds__(256)
void sparsemax_attn_fallback(const float* __restrict__ Q,
                             const float* __restrict__ K,
                             const float* __restrict__ V,
                             float* __restrict__ O) {
    extern __shared__ float smem[];
    const int qtile = blockIdx.x;
    const int bh    = blockIdx.y;
    const int tid   = threadIdx.x;
    const int wave  = tid >> 5;
    const int lane  = tid & 31;
    const int half  = lane >> 4;
    const int l16   = lane & 15;

    const long baseQ = ((long)bh * S_LEN + (long)qtile * QTILE) * D_DIM;
    const long baseK = (long)bh * S_LEN * D_DIM;

    v16bf aq[2];
#pragma unroll
    for (int c = 0; c < 2; ++c)
#pragma unroll
        for (int e = 0; e < 16; ++e)
            aq[c][e] = f2bf(Q[baseQ + (long)l16 * D_DIM + 32 * c + kmap(e, half)] * SCALE);

    for (int i = 0; i < 16; ++i) {
        int kt  = wave * 16 + i;
        int key = kt * 16 + l16;
        v8f acc = {};
#pragma unroll
        for (int c = 0; c < 2; ++c) {
            v16bf bk;
#pragma unroll
            for (int e = 0; e < 16; ++e)
                bk[e] = f2bf(K[baseK + (long)key * D_DIM + 32 * c + kmap(e, half)]);
            acc = __builtin_amdgcn_wmma_f32_16x16x32_bf16(
                      false, aq[c], false, bk, (short)0, acc, false, false);
        }
#pragma unroll
        for (int r = 0; r < 8; ++r)
            smem[(r + 8 * half) * S_LEN + kt * 16 + l16] = acc[r];
    }
    __syncthreads();

#pragma unroll
    for (int rr = 0; rr < 2; ++rr) {
        int m = wave * 2 + rr;
        float* row = smem + m * S_LEN;
        float s = 0.f;
        for (int i = lane; i < S_LEN; i += 32) s += row[i];
        s = wave_sum(s);
        float cnt = (float)S_LEN;
        float tau = (s - 1.f) / cnt;
        for (int it = 0; it < 24; ++it) {
            float s2 = 0.f, c2 = 0.f;
            for (int i = lane; i < S_LEN; i += 32) {
                float z = row[i];
                if (z > tau) { s2 += z; c2 += 1.f; }
            }
            s2 = wave_sum(s2);
            c2 = wave_sum(c2);
            if (c2 == cnt) break;
            cnt = c2;
            tau = (s2 - 1.f) / c2;
        }
        for (int i = lane; i < S_LEN; i += 32) row[i] = fmaxf(row[i] - tau, 0.f);
    }
    __syncthreads();

    v8f acc[4] = { {}, {}, {}, {} };
    for (int j = 0; j < 8; ++j) {
        int keybase = (wave + 8 * j) * 32;
        v16bf ap;
#pragma unroll
        for (int e = 0; e < 16; ++e)
            ap[e] = f2bf(smem[l16 * S_LEN + keybase + kmap(e, half)]);
#pragma unroll
        for (int t = 0; t < 4; ++t) {
            v16bf bv;
#pragma unroll
            for (int e = 0; e < 16; ++e)
                bv[e] = f2bf(V[baseK + (long)(keybase + kmap(e, half)) * D_DIM + t * 16 + l16]);
            acc[t] = __builtin_amdgcn_wmma_f32_16x16x32_bf16(
                         false, ap, false, bv, (short)0, acc[t], false, false);
        }
    }
    __syncthreads();
#pragma unroll
    for (int t = 0; t < 4; ++t)
#pragma unroll
        for (int r = 0; r < 8; ++r)
            smem[((wave * 4 + t) * 8 + r) * 32 + lane] = acc[t][r];
    __syncthreads();

    if (wave < 4) {
        int t = wave;
#pragma unroll
        for (int r = 0; r < 8; ++r) {
            float s = 0.f;
#pragma unroll
            for (int w = 0; w < 8; ++w)
                s += smem[((w * 4 + t) * 8 + r) * 32 + lane];
            long q = (long)bh * S_LEN + (long)qtile * QTILE + (r + 8 * half);
            O[q * D_DIM + t * 16 + l16] = s;
        }
    }
}

extern "C" void kernel_launch(void* const* d_in, const int* in_sizes, int n_in,
                              void* d_out, int out_size, void* d_ws, size_t ws_size,
                              hipStream_t stream) {
    (void)in_sizes; (void)n_in; (void)out_size;
    const float* Q = (const float*)d_in[0];
    const float* K = (const float*)d_in[1];
    const float* V = (const float*)d_in[2];
    float* O = (float*)d_out;

    dim3 grid(S_LEN / QTILE, BH_N);
    dim3 block(256);

    if (d_ws != nullptr && ws_size >= WS_NEEDED) {
        __bf16* Qb = (__bf16*)d_ws;
        __bf16* Kb = Qb + N_ELEM;
        __bf16* Vt = Kb + N_ELEM;
        prep_kernel<<<dim3((unsigned)(N_ELEM / 256)), block, 0, stream>>>(Q, K, V, Qb, Kb, Vt);
        (void)hipFuncSetAttribute((const void*)sparsemax_attn_kernel,
                                  hipFuncAttributeMaxDynamicSharedMemorySize, SMEM_BYTES);
        sparsemax_attn_kernel<<<grid, block, SMEM_BYTES, stream>>>(Qb, Kb, Vt, O);
    } else {
        (void)hipFuncSetAttribute((const void*)sparsemax_attn_fallback,
                                  hipFuncAttributeMaxDynamicSharedMemorySize, SMEM_BYTES);
        sparsemax_attn_fallback<<<grid, block, SMEM_BYTES, stream>>>(Q, K, V, O);
    }
}
